// MultiHeadAttention_69441031241781
// MI455X (gfx1250) — matrix operand order repaired and checked by reading, not compiled, not run on hardware
//
#include <hip/hip_runtime.h>

typedef __bf16 bf16;
typedef __attribute__((ext_vector_type(16))) __bf16 v16bf;
typedef __attribute__((ext_vector_type(8)))  float  v8f;

static constexpr int DMODEL = 1024;
static constexpr int SEQ    = 2048;
static constexpr int NHEAD  = 16;
static constexpr int DK     = 64;
static constexpr int BATCH  = 2;
static constexpr int MTOT   = BATCH * SEQ;   // 4096

static constexpr int LDT = 40;   // LDS row stride (elems) for K=32 tiles (80B, 16B-aligned)
static constexpr int LDK = 72;   // LDS row stride (elems) for K=64 tiles (144B, 16B-aligned)

// gfx1250 async global->LDS copies (ASYNCcnt-tracked, no VGPR round-trip).
#if defined(__has_builtin)
#if __has_builtin(__builtin_amdgcn_global_load_async_to_lds_b128) && \
    __has_builtin(__builtin_amdgcn_s_wait_asynccnt)
#define HAVE_ASYNC_LDS 1
#endif
#endif
#ifndef HAVE_ASYNC_LDS
#define HAVE_ASYNC_LDS 0
#endif

// Probe-derived builtin signature on this toolchain:
//   (v4i32* src, __attribute__((address_space(3))) v4i32* dst, imm, imm)
typedef int v4i32 __attribute__((vector_size(16)));
#define ASRC(p) ((v4i32*)(p))
#define AS3(p)  ((__attribute__((address_space(3))) v4i32*)(p))

// A-operand fragment (16x32 bf16): lane holds row lane%16; per ISA table,
// elems 0..7 -> K = h*8 + j, elems 8..15 -> K = 16 + h*8 + j  (h = lane/16)
__device__ __forceinline__ v16bf frag_a(const bf16* row, int h) {
  v16bf f;
#pragma unroll
  for (int j = 0; j < 8; ++j) f[j] = row[h * 8 + j];
#pragma unroll
  for (int j = 0; j < 8; ++j) f[8 + j] = row[16 + h * 8 + j];
  return f;
}

// B-operand fragment (32x16 bf16): lane holds column lane%16;
// elems 0..7 -> K = h*8 + j, elems 8..15 -> K = 16 + h*8 + j  (h = lane/16)
__device__ __forceinline__ v16bf frag_b(const bf16* row, int h) {
  v16bf f;
#pragma unroll
  for (int j = 0; j < 8; ++j) f[j] = row[h * 8 + j];
#pragma unroll
  for (int j = 0; j < 8; ++j) f[8 + j] = row[16 + h * 8 + j];
  return f;
}

__device__ __forceinline__ v8f wmma_bf16(v16bf a, v16bf b, v8f c) {
  return __builtin_amdgcn_wmma_f32_16x16x32_bf16(false, a, false, b, (short)0, c,
                                                 false, false);
}

// Stage the 128x32 A tile into LDS as bf16. For bf16 inputs this is a pure
// copy -> use the CDNA5 async global->LDS path; fp32 inputs need conversion.
template <typename AT>
__device__ __forceinline__ void stage_A(const AT* __restrict__ A, bf16* lA,
                                        int t, int m0, int k0) {
#if HAVE_ASYNC_LDS
  if constexpr (__is_same(AT, bf16)) {
    // 8KB tile = 512 x 16B chunks over 256 threads -> 2 async b128 copies each
#pragma unroll
    for (int i = 0; i < 2; ++i) {
      const int c   = t + i * 256;
      const int row = c >> 2;
      const int col = (c & 3) * 8;
      __builtin_amdgcn_global_load_async_to_lds_b128(
          ASRC(A + (size_t)(m0 + row) * DMODEL + k0 + col),
          AS3(&lA[row * LDT + col]), 0, 0);
    }
    return;
  }
#endif
#pragma unroll
  for (int i = 0; i < 4; ++i) {
    const int row = (t >> 3) + i * 32;
    const int col = (t & 7) * 4;
    const AT* src = A + (size_t)(m0 + row) * DMODEL + k0 + col;
    bf16* dst = &lA[row * LDT + col];
#pragma unroll
    for (int jj = 0; jj < 4; ++jj) dst[jj] = (bf16)(float)src[jj];
  }
}

// C[M,N] = A[M,K] x W[K,N] + bias ; M=4096, N=K=1024.
// Block tile 128x128, BK=32, 256 threads = 8 waves in a 4(M) x 2(N) grid;
// each wave owns 32x64 = 2x4 WMMA tiles.
template <typename AT, typename OT, bool SPLIT_HEADS>
__global__ __launch_bounds__(256) void gemm_bias_kernel(
    const AT* __restrict__ A, const float* __restrict__ W,
    const float* __restrict__ bias, OT* __restrict__ out) {
  __shared__ __align__(16) bf16 lA[128 * LDT];      // A tile, row-major [m][k]
  __shared__ __align__(16) bf16 lB[128 * LDT];      // W tile transposed [n][k]

  const int t    = threadIdx.x;
  const int w    = t >> 5;
  const int lane = t & 31;
  const int r    = lane & 15;
  const int h    = lane >> 4;
  const int m0   = blockIdx.y * 128;
  const int n0   = blockIdx.x * 128;
  const int wm   = (w & 3) * 32;
  const int wn   = (w >> 2) * 64;

  v8f acc[2][4] = {};

  for (int k0 = 0; k0 < DMODEL; k0 += 32) {
    // prefetch next tiles into GL2 while staging this one
    if (k0 + 32 < DMODEL) {
      __builtin_prefetch(A + (size_t)(m0 + (t >> 3)) * DMODEL + k0 + 32, 0, 0);
      __builtin_prefetch(W + (size_t)(k0 + 32 + (t >> 5)) * DMODEL + n0, 0, 0);
    }
    stage_A<AT>(A, lA, t, m0, k0);
    // stage W tile 32x128 (fp32), transposed into lB[n][k] as bf16
#pragma unroll
    for (int i = 0; i < 4; ++i) {
      const int kr = (t >> 5) + i * 8;
      const int nc = (t & 31) * 4;
      const float* src = W + (size_t)(k0 + kr) * DMODEL + n0 + nc;
#pragma unroll
      for (int jj = 0; jj < 4; ++jj) lB[(nc + jj) * LDT + kr] = (bf16)src[jj];
    }
#if HAVE_ASYNC_LDS
    if constexpr (__is_same(AT, bf16)) __builtin_amdgcn_s_wait_asynccnt(0);
#endif
    __syncthreads();

    v16bf af[2], bfr[4];
#pragma unroll
    for (int i = 0; i < 2; ++i) af[i] = frag_a(&lA[(wm + i * 16 + r) * LDT], h);
#pragma unroll
    for (int jn = 0; jn < 4; ++jn)
      bfr[jn] = frag_b(&lB[(wn + jn * 16 + r) * LDT], h);
#pragma unroll
    for (int i = 0; i < 2; ++i)
#pragma unroll
      for (int jn = 0; jn < 4; ++jn)
        acc[i][jn] = wmma_bf16(af[i], bfr[jn], acc[i][jn]);
    __syncthreads();
  }

  // epilogue: C layout is M = e + 8*h, N = lane%16
#pragma unroll
  for (int i = 0; i < 2; ++i) {
#pragma unroll
    for (int jn = 0; jn < 4; ++jn) {
      const int ncol = n0 + wn + jn * 16 + r;
      const float bv = bias[ncol];
#pragma unroll
      for (int e = 0; e < 8; ++e) {
        const int mm    = m0 + wm + i * 16 + e + 8 * h;
        const float val = acc[i][jn][e] + bv;
        if (SPLIT_HEADS) {
          const int hd = ncol >> 6, d = ncol & 63;
          const int bb = mm >> 11, ss = mm & 2047;
          out[(((size_t)bb * NHEAD + hd) * SEQ + ss) * DK + d] = (OT)val;
        } else {
          out[(size_t)mm * DMODEL + ncol] = (OT)val;
        }
      }
    }
  }
}

// Flash attention: one workgroup = one (b,h) and 64 query rows; 4 waves, each
// owns 16 rows. Key blocks of 64, online softmax, score matrix never hits HBM.
__global__ __launch_bounds__(128) void flash_attn_kernel(
    const bf16* __restrict__ qh, const bf16* __restrict__ kh,
    const bf16* __restrict__ vh, bf16* __restrict__ ctx) {
  __shared__ __align__(16) bf16 lK[64 * LDK];        // [key][d]  (B-op rows for scores)
  __shared__ __align__(16) bf16 lVt[64 * LDK];       // [d][key]  (B-op rows for P*V)
  __shared__ __align__(16) bf16 lP[4 * 16 * LDK];    // per-wave P tile [row][key]

  const int t    = threadIdx.x;
  const int w    = t >> 5;
  const int lane = t & 31;
  const int r    = lane & 15;
  const int h    = lane >> 4;
  const int bh   = blockIdx.y;                 // b*H + head
  const int q0   = blockIdx.x * 64 + w * 16;   // this wave's first query row

  const bf16* qp = qh + ((size_t)bh * SEQ + q0) * DK;
  const bf16* kp = kh + (size_t)bh * SEQ * DK;
  const bf16* vp = vh + (size_t)bh * SEQ * DK;
  bf16* lPw = lP + w * 16 * LDK;

  v16bf qf[2];
  qf[0] = frag_a(qp + (size_t)r * DK, h);
  qf[1] = frag_a(qp + (size_t)r * DK + 32, h);

  v8f acc[4] = {};
  float m[8], l[8];
#pragma unroll
  for (int e = 0; e < 8; ++e) { m[e] = -3.0e38f; l[e] = 0.f; }
  const float scale = 0.125f;  // 1/sqrt(64)

  for (int j = 0; j < SEQ / 64; ++j) {
    __syncthreads();
    {
      const int row = t >> 1;
      const int c0  = (t & 1) * 32;
      // K tile: pure bf16 copy -> async global->LDS (4 x b128 per thread)
#if HAVE_ASYNC_LDS
#pragma unroll
      for (int q = 0; q < 4; ++q)
        __builtin_amdgcn_global_load_async_to_lds_b128(
            ASRC(kp + (size_t)(j * 64 + row) * DK + c0 + q * 8),
            AS3(&lK[row * LDK + c0 + q * 8]), 0, 0);
#else
      const uint4* ks = (const uint4*)(kp + (size_t)(j * 64 + row) * DK + c0);
      uint4* kd = (uint4*)&lK[row * LDK + c0];
#pragma unroll
      for (int q = 0; q < 4; ++q) kd[q] = ks[q];
#endif
      // V tile: transpose while staging (needs VGPR path)
      const bf16* vs = vp + (size_t)(j * 64 + row) * DK + c0;
#pragma unroll
      for (int d = 0; d < 32; ++d) lVt[(c0 + d) * LDK + row] = vs[d];
      // prefetch next key block into GL2
      if (j + 1 < SEQ / 64) {
        __builtin_prefetch(kp + (size_t)((j + 1) * 64 + row) * DK, 0, 0);
        __builtin_prefetch(vp + (size_t)((j + 1) * 64 + row) * DK, 0, 0);
      }
    }
#if HAVE_ASYNC_LDS
    __builtin_amdgcn_s_wait_asynccnt(0);
#endif
    __syncthreads();

    // scores: S[16 x 64] = q(16x64) x K^T ; B column = key row of lK (contiguous)
    v8f s[4];
#pragma unroll
    for (int nt = 0; nt < 4; ++nt) {
      v8f c = {};
      c = wmma_bf16(qf[0], frag_b(&lK[(nt * 16 + r) * LDK], h), c);
      c = wmma_bf16(qf[1], frag_b(&lK[(nt * 16 + r) * LDK + 32], h), c);
      s[nt] = c;
    }

    // online softmax; row M = e + 8*h lives across 16 lanes of one half
    float mloc[8];
#pragma unroll
    for (int e = 0; e < 8; ++e) {
      float v = fmaxf(fmaxf(s[0][e], s[1][e]), fmaxf(s[2][e], s[3][e]));
      mloc[e] = v * scale;
    }
#pragma unroll
    for (int off = 1; off < 16; off <<= 1)
#pragma unroll
      for (int e = 0; e < 8; ++e)
        mloc[e] = fmaxf(mloc[e], __shfl_xor(mloc[e], off, 32));

    float alpha[8], rs[8];
#pragma unroll
    for (int e = 0; e < 8; ++e) {
      const float mn = fmaxf(m[e], mloc[e]);
      alpha[e] = __expf(m[e] - mn);
      m[e]     = mn;
      rs[e]    = 0.f;
    }
#pragma unroll
    for (int nt = 0; nt < 4; ++nt)
#pragma unroll
      for (int e = 0; e < 8; ++e) {
        const float p = __expf(s[nt][e] * scale - m[e]);
        rs[e] += p;
        lPw[(e + 8 * h) * LDK + nt * 16 + r] = (bf16)p;  // C-layout -> row-major
      }
#pragma unroll
    for (int off = 1; off < 16; off <<= 1)
#pragma unroll
      for (int e = 0; e < 8; ++e) rs[e] += __shfl_xor(rs[e], off, 32);
#pragma unroll
    for (int e = 0; e < 8; ++e) l[e] = l[e] * alpha[e] + rs[e];
#pragma unroll
    for (int dt = 0; dt < 4; ++dt)
#pragma unroll
      for (int e = 0; e < 8; ++e) acc[dt][e] *= alpha[e];

    // O += P(16x64) x V(64x64); B column = dk row of lVt (contiguous)
#pragma unroll
    for (int kk = 0; kk < 2; ++kk) {
      v16bf a = frag_a(&lPw[r * LDK + kk * 32], h);
#pragma unroll
      for (int dt = 0; dt < 4; ++dt)
        acc[dt] = wmma_bf16(a, frag_b(&lVt[(dt * 16 + r) * LDK + kk * 32], h),
                            acc[dt]);
    }
  }

  // finalize and write combined-head ctx [B*S, D_MODEL] (bf16)
  const int bb = bh >> 4;
  const int hd = bh & 15;
#pragma unroll
  for (int e = 0; e < 8; ++e) {
    const float inv = 1.f / l[e];
    const int qrow  = q0 + e + 8 * h;
    const size_t rowg = (size_t)(bb * SEQ + qrow) * DMODEL + hd * DK;
#pragma unroll
    for (int dt = 0; dt < 4; ++dt)
      ctx[rowg + dt * 16 + r] = (bf16)(acc[dt][e] * inv);
  }
}

extern "C" void kernel_launch(void* const* d_in, const int* in_sizes, int n_in,
                              void* d_out, int out_size, void* d_ws,
                              size_t ws_size, hipStream_t stream) {
  (void)in_sizes; (void)n_in; (void)out_size; (void)ws_size;
  const float* Q  = (const float*)d_in[0];
  const float* K  = (const float*)d_in[1];
  const float* V  = (const float*)d_in[2];
  const float* Wq = (const float*)d_in[3];
  const float* bq = (const float*)d_in[4];
  const float* Wk = (const float*)d_in[5];
  const float* bk = (const float*)d_in[6];
  const float* Wv = (const float*)d_in[7];
  const float* bv = (const float*)d_in[8];
  const float* Wo = (const float*)d_in[9];
  const float* bo = (const float*)d_in[10];
  float* out = (float*)d_out;

  const size_t elems = (size_t)MTOT * DMODEL;   // 4.19M elems, 8MB bf16 each
  bf16* qh = (bf16*)d_ws;
  bf16* kh = qh + elems;
  bf16* vh = kh + elems;
  bf16* ct = vh + elems;                         // total 32MB workspace

  dim3 gg(DMODEL / 128, MTOT / 128), bg(256);
  hipLaunchKernelGGL((gemm_bias_kernel<float, bf16, true>), gg, bg, 0, stream,
                     Q, Wq, bq, qh);
  hipLaunchKernelGGL((gemm_bias_kernel<float, bf16, true>), gg, bg, 0, stream,
                     K, Wk, bk, kh);
  hipLaunchKernelGGL((gemm_bias_kernel<float, bf16, true>), gg, bg, 0, stream,
                     V, Wv, bv, vh);
  hipLaunchKernelGGL(flash_attn_kernel, dim3(SEQ / 64, BATCH * NHEAD),
                     dim3(128), 0, stream, qh, kh, vh, ct);
  hipLaunchKernelGGL((gemm_bias_kernel<bf16, float, false>), gg, bg, 0, stream,
                     ct, Wo, bo, out);
}
